// MoE_70394513981761
// MI455X (gfx1250) — matrix operand order repaired and checked by reading, not compiled, not run on hardware
//
#include <hip/hip_runtime.h>
#include <math.h>

// ---------------- problem constants ----------------
constexpr int T_TOK    = 8 * 196;   // 1568 tokens
constexpr int DIM      = 768;
constexpr int NEXP     = 16;
constexpr int TOPK     = 2;
constexpr int MOE_I    = 2048;
constexpr int SHARED_I = 4096;

// ---------------- GEMM tiling ----------------
constexpr int BM  = 64;             // token rows per block
constexpr int BN  = 128;            // output cols per block
constexpr int BK  = 64;             // K slice per LDS stage (2 WMMA k-steps)
constexpr int LDK = BK + 8;         // padded LDS row stride (bank spread)
constexpr int MTR = (TOPK * T_TOK + BM - 1) / BM;  // max row tiles per expert = 49

typedef __attribute__((ext_vector_type(16))) __bf16 bf16x16;
typedef __attribute__((ext_vector_type(8)))  __bf16 bf16x8;
typedef __attribute__((ext_vector_type(4)))  __bf16 bf16x4;
typedef __attribute__((ext_vector_type(8)))  float  f32x8;
typedef int v4i __attribute__((vector_size(16)));

#ifndef __has_builtin
#define __has_builtin(x) 0
#endif
#if __has_builtin(__builtin_amdgcn_global_load_async_to_lds_b128)
#define HAVE_ASYNC_LDS 1
#else
#define HAVE_ASYNC_LDS 0
#endif

// 16-byte global -> LDS copy: async (ASYNCcnt) when available, sync fallback.
__device__ __forceinline__ void copy16_g2l(const __bf16* g, __bf16* l) {
#if HAVE_ASYNC_LDS
    __builtin_amdgcn_global_load_async_to_lds_b128(
        (__attribute__((address_space(1))) v4i*)g,
        (__attribute__((address_space(3))) v4i*)l, 0, 0);
#else
    *(uint4*)l = *(const uint4*)g;
#endif
}
__device__ __forceinline__ void wait_async_all() {
#if HAVE_ASYNC_LDS
#if __has_builtin(__builtin_amdgcn_s_wait_asynccnt)
    __builtin_amdgcn_s_wait_asynccnt(0);
#else
    asm volatile("s_wait_asynccnt 0x0" ::: "memory");
#endif
#endif
}

// ---- WMMA fragment loaders (layouts per CDNA5 ISA §7.12.2) ----
// A 16x32 bf16: lane<16 -> M=lane, K={0..7,16..23}; lane>=16 -> M=lane-16, K={8..15,24..31}
__device__ __forceinline__ bf16x16 load_frag_a(const __bf16* rowk, int hi) {
    const __bf16* q = rowk + hi * 8;
    bf16x8 lo = *(const bf16x8*)q;
    bf16x8 hh = *(const bf16x8*)(q + 16);
    return __builtin_shufflevector(lo, hh, 0,1,2,3,4,5,6,7,8,9,10,11,12,13,14,15);
}
// B 32x16 bf16: lane<16 -> N=lane, K={0..7,16..23}; lane>=16 -> N=lane-16, K={8..15,24..31}
__device__ __forceinline__ bf16x16 load_frag_b(const __bf16* rowk, int hi) {
    const __bf16* q = rowk + hi * 8;
    bf16x8 lo = *(const bf16x8*)q;
    bf16x8 hh = *(const bf16x8*)(q + 16);
    return __builtin_shufflevector(lo, hh, 0,1,2,3,4,5,6,7,8,9,10,11,12,13,14,15);
}

// ---------------- kernel: fp32 -> bf16 bulk convert (weights stay L2-resident) ----------------
__global__ __launch_bounds__(256)
void cvt_bf16(const float* __restrict__ src, __bf16* __restrict__ dst, int n) {
    int i = (blockIdx.x * 256 + threadIdx.x) * 4;
    if (i < n) {
        float4 f = *(const float4*)(src + i);
        bf16x4 o;
        o[0] = (__bf16)f.x; o[1] = (__bf16)f.y; o[2] = (__bf16)f.z; o[3] = (__bf16)f.w;
        *(bf16x4*)(dst + i) = o;
    }
}

// ---------------- kernel: x -> bf16, zero counts ----------------
__global__ void moe_init(const float* __restrict__ x, __bf16* __restrict__ xbf,
                         int n, int* __restrict__ counts) {
    int i = blockIdx.x * 256 + threadIdx.x;
    if (i < n) xbf[i] = (__bf16)x[i];
    if (i < NEXP) counts[i] = 0;
}

// ---------------- kernel: gate (softmax + top-2) ----------------
__global__ __launch_bounds__(256)
void gate_kernel(const float* __restrict__ x, const float* __restrict__ gw,
                 int* __restrict__ topi, float* __restrict__ topw,
                 int* __restrict__ counts) {
    int t = blockIdx.x;
    __shared__ float xs[DIM];
    __shared__ float sc[NEXP];
    int tid = threadIdx.x;
    for (int i = tid; i < DIM; i += 256) xs[i] = x[(size_t)t * DIM + i];
    __syncthreads();
    int wid = tid >> 5, lane = tid & 31;
    for (int e = wid; e < NEXP; e += 8) {
        float s = 0.f;
        for (int i = lane; i < DIM; i += 32) s += xs[i] * gw[(size_t)e * DIM + i];
        #pragma unroll
        for (int o = 16; o > 0; o >>= 1) s += __shfl_xor(s, o, 32);
        if (lane == 0) sc[e] = s;
    }
    __syncthreads();
    if (tid == 0) {
        float mx = sc[0];
        for (int e = 1; e < NEXP; e++) mx = fmaxf(mx, sc[e]);
        float den = 0.f, p[NEXP];
        for (int e = 0; e < NEXP; e++) { p[e] = __expf(sc[e] - mx); den += p[e]; }
        float inv = 1.f / den;
        int i1 = 0; float v1 = -1.f;
        for (int e = 0; e < NEXP; e++) if (p[e] > v1) { v1 = p[e]; i1 = e; }
        int i2 = -1; float v2 = -1.f;
        for (int e = 0; e < NEXP; e++) if (e != i1 && p[e] > v2) { v2 = p[e]; i2 = e; }
        topi[2 * t]     = i1;  topw[2 * t]     = v1 * inv;
        topi[2 * t + 1] = i2;  topw[2 * t + 1] = v2 * inv;
        atomicAdd(&counts[i1], 1);
        atomicAdd(&counts[i2], 1);
    }
}

// ---------------- kernel: exclusive scan over 16 counts ----------------
__global__ void scan_kernel(const int* __restrict__ counts,
                            int* __restrict__ offsets, int* __restrict__ cursor) {
    if (threadIdx.x == 0) {
        int s = 0;
        for (int e = 0; e < NEXP; e++) { offsets[e] = s; cursor[e] = s; s += counts[e]; }
        offsets[NEXP] = s;
    }
}

// ---------------- kernel: scatter (token,k) -> expert-grouped rows ----------------
__global__ void scatter_kernel(const int* __restrict__ topi, const float* __restrict__ topw,
                               int* __restrict__ cursor, int* __restrict__ rowTok,
                               int* __restrict__ rowSlot, float* __restrict__ rowW) {
    int t = blockIdx.x * 256 + threadIdx.x;
    if (t >= T_TOK) return;
    #pragma unroll
    for (int k = 0; k < TOPK; k++) {
        int e = topi[2 * t + k];
        int pos = atomicAdd(&cursor[e], 1);
        rowTok[pos]  = t;
        rowSlot[pos] = 2 * t + k;
        rowW[pos]    = topw[2 * t + k];
    }
}

// ---------------- up-projection: dual-B SwiGLU GEMM (bf16 weights, async+double buffer) ----------------
template <bool ROUTED>
__global__ __launch_bounds__(256)
void gemm_up(const __bf16* __restrict__ xbf,
             const __bf16* __restrict__ w1, const __bf16* __restrict__ w3,
             __bf16* __restrict__ hout,
             const int* __restrict__ offsets, const int* __restrict__ rowTok,
             const float* __restrict__ rowW, int NI) {
    __shared__ __bf16 As [2][BM * LDK];
    __shared__ __bf16 B1s[2][BN * LDK];
    __shared__ __bf16 B3s[2][BN * LDK];
    __shared__ float  rwS[BM];

    int tid = threadIdx.x;
    int e = 0, pbase, rows;
    if (ROUTED) {
        e = blockIdx.x / MTR;
        int mt = blockIdx.x % MTR;
        int beg = offsets[e], end = offsets[e + 1];
        pbase = beg + mt * BM;
        rows = end - pbase;
        if (rows <= 0) return;
        if (rows > BM) rows = BM;
    } else {
        pbase = blockIdx.x * BM;
        rows = T_TOK - pbase;
        if (rows > BM) rows = BM;
    }
    int nbase = blockIdx.y * BN;
    const __bf16* w1e = w1 + (ROUTED ? (size_t)e * NI * DIM : 0);
    const __bf16* w3e = w3 + (ROUTED ? (size_t)e * NI * DIM : 0);

    if (ROUTED && tid < BM) rwS[tid] = (tid < rows) ? rowW[pbase + tid] : 0.f;

    int lane = tid & 31, wid = tid >> 5;
    int wm = wid >> 2, wn = wid & 3;      // 2 x 4 wave grid
    int hi = lane >> 4, lm = lane & 15;

    const f32x8 vzero = {0.f,0.f,0.f,0.f,0.f,0.f,0.f,0.f};
    f32x8 acc1[2][2], acc3[2][2];
    #pragma unroll
    for (int i = 0; i < 2; i++)
        #pragma unroll
        for (int j = 0; j < 2; j++) { acc1[i][j] = vzero; acc3[i][j] = vzero; }

    auto issue = [&](int s, int buf) {
        int k0 = s * BK;
        for (int c = tid; c < BM * (BK / 8); c += 256) {
            int row = c >> 3;
            int k8  = (c & 7) << 3;
            __bf16* dst = &As[buf][row * LDK + k8];
            if (row < rows) {
                int tok = ROUTED ? rowTok[pbase + row] : (pbase + row);
                copy16_g2l(xbf + (size_t)tok * DIM + (k0 + k8), dst);
            } else {
                *(uint4*)dst = make_uint4(0u, 0u, 0u, 0u);
            }
        }
        for (int c = tid; c < BN * (BK / 8); c += 256) {
            int row = c >> 3;
            int k8  = (c & 7) << 3;
            size_t off = (size_t)(nbase + row) * DIM + (k0 + k8);
            copy16_g2l(w1e + off, &B1s[buf][row * LDK + k8]);
            copy16_g2l(w3e + off, &B3s[buf][row * LDK + k8]);
        }
    };

    constexpr int NSTAGE = DIM / BK;   // 12
    issue(0, 0);
    for (int s = 0; s < NSTAGE; ++s) {
        wait_async_all();              // my stage-s transfers landed
        __syncthreads();               // everyone's landed; prior reads retired
        if (s + 1 < NSTAGE) issue(s + 1, (s + 1) & 1);
        const __bf16* Ab  = As [s & 1];
        const __bf16* B1b = B1s[s & 1];
        const __bf16* B3b = B3s[s & 1];
        #pragma unroll
        for (int kk = 0; kk < BK; kk += 32) {
            bf16x16 af[2], b1f[2], b3f[2];
            #pragma unroll
            for (int st = 0; st < 2; st++) {
                af[st]  = load_frag_a(&Ab [(wm * 32 + st * 16 + lm) * LDK + kk], hi);
                b1f[st] = load_frag_b(&B1b[(wn * 32 + st * 16 + lm) * LDK + kk], hi);
                b3f[st] = load_frag_b(&B3b[(wn * 32 + st * 16 + lm) * LDK + kk], hi);
            }
            #pragma unroll
            for (int sm = 0; sm < 2; sm++)
                #pragma unroll
                for (int sn = 0; sn < 2; sn++) {
                    acc1[sm][sn] = __builtin_amdgcn_wmma_f32_16x16x32_bf16(
                        false, af[sm], false, b1f[sn], (short)0, acc1[sm][sn], false, false);
                    acc3[sm][sn] = __builtin_amdgcn_wmma_f32_16x16x32_bf16(
                        false, af[sm], false, b3f[sn], (short)0, acc3[sm][sn], false, false);
                }
        }
    }
    // epilogue: silu(a1)*a3 (* route weight) -> bf16
    #pragma unroll
    for (int sm = 0; sm < 2; sm++)
        #pragma unroll
        for (int sn = 0; sn < 2; sn++)
            #pragma unroll
            for (int j = 0; j < 8; j++) {
                int row = wm * 32 + sm * 16 + hi * 8 + j;
                if (row < rows) {
                    float v1 = acc1[sm][sn][j];
                    float v3 = acc3[sm][sn][j];
                    float hv = v1 * __frcp_rn(1.f + __expf(-v1)) * v3;
                    if (ROUTED) hv *= rwS[row];
                    int ncol = nbase + wn * 32 + sn * 16 + lm;
                    hout[(size_t)(pbase + row) * NI + ncol] = (__bf16)hv;
                }
            }
}

// ---------------- down-projection GEMM (bf16 weights, async+double buffer) ----------------
// ROUTED: ycontrib[slot, :768] = H @ W2^T ; SHARED: out = H @ SW2^T + ycontrib sums
template <bool ROUTED>
__global__ __launch_bounds__(256)
void gemm_down(const __bf16* __restrict__ hin, const __bf16* __restrict__ w2,
               float* __restrict__ outp, const float* __restrict__ ycontrib,
               const int* __restrict__ offsets, const int* __restrict__ rowSlot, int K) {
    __shared__ __bf16 As[2][BM * LDK];
    __shared__ __bf16 Bs[2][BN * LDK];
    __shared__ int    slotS[BM];

    int tid = threadIdx.x;
    int e = 0, pbase, rows;
    if (ROUTED) {
        e = blockIdx.x / MTR;
        int mt = blockIdx.x % MTR;
        int beg = offsets[e], end = offsets[e + 1];
        pbase = beg + mt * BM;
        rows = end - pbase;
        if (rows <= 0) return;
        if (rows > BM) rows = BM;
    } else {
        pbase = blockIdx.x * BM;
        rows = T_TOK - pbase;
        if (rows > BM) rows = BM;
    }
    int nbase = blockIdx.y * BN;
    const __bf16* w2e = w2 + (ROUTED ? (size_t)e * DIM * K : 0);

    if (ROUTED && tid < BM) slotS[tid] = (tid < rows) ? rowSlot[pbase + tid] : 0;

    int lane = tid & 31, wid = tid >> 5;
    int wm = wid >> 2, wn = wid & 3;
    int hi = lane >> 4, lm = lane & 15;

    const f32x8 vzero = {0.f,0.f,0.f,0.f,0.f,0.f,0.f,0.f};
    f32x8 acc[2][2];
    #pragma unroll
    for (int i = 0; i < 2; i++)
        #pragma unroll
        for (int j = 0; j < 2; j++) acc[i][j] = vzero;

    auto issue = [&](int s, int buf) {
        int k0 = s * BK;
        for (int c = tid; c < BM * (BK / 8); c += 256) {
            int row = c >> 3;
            int k8  = (c & 7) << 3;
            __bf16* dst = &As[buf][row * LDK + k8];
            if (row < rows)
                copy16_g2l(hin + (size_t)(pbase + row) * K + (k0 + k8), dst);
            else
                *(uint4*)dst = make_uint4(0u, 0u, 0u, 0u);
        }
        for (int c = tid; c < BN * (BK / 8); c += 256) {
            int row = c >> 3;
            int k8  = (c & 7) << 3;
            copy16_g2l(w2e + (size_t)(nbase + row) * K + (k0 + k8),
                       &Bs[buf][row * LDK + k8]);
        }
    };

    int nst = K / BK;
    issue(0, 0);
    for (int s = 0; s < nst; ++s) {
        wait_async_all();
        __syncthreads();
        if (s + 1 < nst) issue(s + 1, (s + 1) & 1);
        const __bf16* Ab = As[s & 1];
        const __bf16* Bb = Bs[s & 1];
        #pragma unroll
        for (int kk = 0; kk < BK; kk += 32) {
            bf16x16 af[2], bf[2];
            #pragma unroll
            for (int st = 0; st < 2; st++) {
                af[st] = load_frag_a(&Ab[(wm * 32 + st * 16 + lm) * LDK + kk], hi);
                bf[st] = load_frag_b(&Bb[(wn * 32 + st * 16 + lm) * LDK + kk], hi);
            }
            #pragma unroll
            for (int sm = 0; sm < 2; sm++)
                #pragma unroll
                for (int sn = 0; sn < 2; sn++)
                    acc[sm][sn] = __builtin_amdgcn_wmma_f32_16x16x32_bf16(
                        false, af[sm], false, bf[sn], (short)0, acc[sm][sn], false, false);
        }
    }
    #pragma unroll
    for (int sm = 0; sm < 2; sm++)
        #pragma unroll
        for (int sn = 0; sn < 2; sn++)
            #pragma unroll
            for (int j = 0; j < 8; j++) {
                int row = wm * 32 + sm * 16 + hi * 8 + j;
                if (row < rows) {
                    int ncol = nbase + wn * 32 + sn * 16 + lm;
                    float v = acc[sm][sn][j];
                    if (ROUTED) {
                        outp[(size_t)slotS[row] * DIM + ncol] = v;
                    } else {
                        int t = pbase + row;
                        outp[(size_t)t * DIM + ncol] =
                            v + ycontrib[(size_t)(2 * t) * DIM + ncol]
                              + ycontrib[(size_t)(2 * t + 1) * DIM + ncol];
                    }
                }
            }
}

// ---------------- host launcher ----------------
extern "C" void kernel_launch(void* const* d_in, const int* in_sizes, int n_in,
                              void* d_out, int out_size, void* d_ws, size_t ws_size,
                              hipStream_t stream) {
    (void)in_sizes; (void)n_in; (void)out_size; (void)ws_size;
    const float* x   = (const float*)d_in[0];
    const float* gw  = (const float*)d_in[1];
    const float* w1  = (const float*)d_in[2];
    const float* w2  = (const float*)d_in[3];
    const float* w3  = (const float*)d_in[4];
    const float* sw1 = (const float*)d_in[5];
    const float* sw2 = (const float*)d_in[6];
    const float* sw3 = (const float*)d_in[7];
    float* out = (float*)d_out;

    char* p = (char*)d_ws;
    auto alloc = [&](size_t b) -> char* {
        char* r = p;
        p += (b + 255) & ~(size_t)255;
        return r;
    };
    float*  topw    = (float*) alloc((size_t)2 * T_TOK * 4);
    int*    topi    = (int*)   alloc((size_t)2 * T_TOK * 4);
    int*    counts  = (int*)   alloc(64);
    int*    offsets = (int*)   alloc(68);
    int*    cursor  = (int*)   alloc(64);
    int*    rowTok  = (int*)   alloc((size_t)2 * T_TOK * 4);
    int*    rowSlot = (int*)   alloc((size_t)2 * T_TOK * 4);
    float*  rowW    = (float*) alloc((size_t)2 * T_TOK * 4);
    __bf16* xbf     = (__bf16*)alloc((size_t)T_TOK * DIM * 2);
    __bf16* h_rout  = (__bf16*)alloc((size_t)2 * T_TOK * MOE_I * 2);
    __bf16* h_sh    = (__bf16*)alloc((size_t)T_TOK * SHARED_I * 2);
    float*  ycon    = (float*) alloc((size_t)2 * T_TOK * DIM * 4);
    // bf16 weight mirrors: 151 MB routed + 19 MB shared -> L2-resident re-reads
    const size_t NW_R = (size_t)NEXP * MOE_I * DIM;       // per routed tensor
    const size_t NW_S = (size_t)SHARED_I * DIM;           // per shared tensor
    __bf16* w1b  = (__bf16*)alloc(NW_R * 2);
    __bf16* w2b  = (__bf16*)alloc(NW_R * 2);
    __bf16* w3b  = (__bf16*)alloc(NW_R * 2);
    __bf16* sw1b = (__bf16*)alloc(NW_S * 2);
    __bf16* sw2b = (__bf16*)alloc(NW_S * 2);
    __bf16* sw3b = (__bf16*)alloc(NW_S * 2);

    auto cvt = [&](const float* s, __bf16* d, size_t n) {
        cvt_bf16<<<(unsigned)((n / 4 + 255) / 256), 256, 0, stream>>>(s, d, (int)n);
    };
    cvt(w1, w1b, NW_R);  cvt(w2, w2b, NW_R);  cvt(w3, w3b, NW_R);
    cvt(sw1, sw1b, NW_S); cvt(sw2, sw2b, NW_S); cvt(sw3, sw3b, NW_S);

    int nx = T_TOK * DIM;
    moe_init<<<(nx + 255) / 256, 256, 0, stream>>>(x, xbf, nx, counts);
    gate_kernel<<<T_TOK, 256, 0, stream>>>(x, gw, topi, topw, counts);
    scan_kernel<<<1, 32, 0, stream>>>(counts, offsets, cursor);
    scatter_kernel<<<(T_TOK + 255) / 256, 256, 0, stream>>>(topi, topw, cursor,
                                                            rowTok, rowSlot, rowW);
    gemm_up<true><<<dim3(NEXP * MTR, MOE_I / BN), 256, 0, stream>>>(
        xbf, w1b, w3b, h_rout, offsets, rowTok, rowW, MOE_I);
    gemm_up<false><<<dim3((T_TOK + BM - 1) / BM, SHARED_I / BN), 256, 0, stream>>>(
        xbf, sw1b, sw3b, h_sh, nullptr, nullptr, nullptr, SHARED_I);
    gemm_down<true><<<dim3(NEXP * MTR, DIM / BN), 256, 0, stream>>>(
        h_rout, w2b, ycon, nullptr, offsets, rowSlot, MOE_I);
    gemm_down<false><<<dim3((T_TOK + BM - 1) / BM, DIM / BN), 256, 0, stream>>>(
        h_sh, sw2b, out, ycon, nullptr, nullptr, SHARED_I);
}
